// LiPSRDFMAE_29145648070879
// MI455X (gfx1250) — compile-verified, matrix operand order checked by reading, not run on hardware
//
#include <hip/hip_runtime.h>
#include <hip/hip_bf16.h>
#include <math.h>

// ---------------- problem constants (from the reference) ----------------
#define T_      1000
#define R_      16
#define N_      83
#define NB      600
#define FPB     10            // frames per block in the histogram kernel
#define NPAIR   3403          // N*(N-1)/2
#define NF      249           // floats per frame (N*3)
#define FSTRIDE 252           // padded floats per frame in LDS
#define B0F     1e-6f
#define XLIMF   6.0f

typedef __attribute__((ext_vector_type(2)))  float    v2f;
typedef __attribute__((ext_vector_type(8)))  float    v8f;
typedef __attribute__((ext_vector_type(16))) _Float16 v16h;

// ---------------- feature probes (compile-time) ----------------
#if defined(__has_builtin)
#  if __has_builtin(__builtin_amdgcn_global_load_async_to_lds_b32) && \
      __has_builtin(__builtin_amdgcn_s_wait_asynccnt)
#    define USE_ASYNC_LDS 1
#  endif
#  if __has_builtin(__builtin_amdgcn_wmma_f32_16x16x4_f32)
#    define HAVE_WMMA_F32X4 1
#  endif
#endif

#define AS1 __attribute__((address_space(1)))
#define AS3 __attribute__((address_space(3)))

// pairs whose first index is < i  :  S(i) = i*(2N-1-i)/2
__device__ __forceinline__ int pairS(int i) { return (i * (2 * N_ - 1 - i)) >> 1; }

// ---------------- k0: zero the global per-replica histograms ----------------
__global__ void k0_zero(unsigned int* __restrict__ p, int n) {
    int i = blockIdx.x * blockDim.x + threadIdx.x;
    if (i < n) p[i] = 0u;
}

// ---------------- k1: LDS-privatized pair-distance histogram ----------------
__global__ __launch_bounds__(256) void k1_hist(const float* __restrict__ pos,
                                               const float* __restrict__ cell,
                                               unsigned int* __restrict__ ghist) {
    __shared__ unsigned int shh[NB];
    __shared__ float        shp[FPB * FSTRIDE];

    const int tid = threadIdx.x;
    const int r   = blockIdx.x;          // replica
    const int t0  = blockIdx.y * FPB;    // first frame handled by this block

    for (int b = tid; b < NB; b += 256) shh[b] = 0u;

    // prefetch the first element this thread will stage (global_prefetch_b8)
    if (tid < FPB * NF) {
        int f = tid / NF, m = tid - f * NF;
        __builtin_prefetch(pos + ((size_t)((t0 + f) * R_ + r)) * NF + m, 0, 1);
    }

    // ---- stage FPB frames of positions into LDS ----
#ifdef USE_ASYNC_LDS
    for (int q = tid; q < FPB * NF; q += 256) {
        int f = q / NF, m = q - f * NF;
        const float* s = pos + ((size_t)((t0 + f) * R_ + r)) * NF + m;
        __builtin_amdgcn_global_load_async_to_lds_b32(
            (AS1 int*)s, (AS3 int*)&shp[f * FSTRIDE + m], 0, 0);
    }
    __builtin_amdgcn_s_wait_asynccnt(0);
    __syncthreads();
#else
    for (int q = tid; q < FPB * NF; q += 256) {
        int f = q / NF, m = q - f * NF;
        shp[f * FSTRIDE + m] = pos[((size_t)((t0 + f) * R_ + r)) * NF + m];
    }
    __syncthreads();
#endif

    // diagonal cell (reference uses 10*I); min-image per component
    const float Lx = cell[0], Ly = cell[4], Lz = cell[8];
    const float iLx = 1.0f / Lx, iLy = 1.0f / Ly, iLz = 1.0f / Lz;
    const float INV_W = (float)NB / (XLIMF - B0F);

    for (int p = tid; p < NPAIR; p += 256) {
        // decode upper-triangular (i,j) from linear index p
        float fi = (165.0f - sqrtf((float)(165 * 165 - 8 * p))) * 0.5f;
        int i = (int)fi;
        if (i < 0) i = 0;
        if (i > N_ - 2) i = N_ - 2;
        while (i < N_ - 2 && pairS(i + 1) <= p) ++i;
        while (i > 0 && pairS(i) > p) --i;
        const int j  = i + 1 + (p - pairS(i));
        const int bi = 3 * i, bj = 3 * j;

        for (int f = 0; f < FPB; ++f) {
            const float* P = shp + f * FSTRIDE;
            float dx = P[bi + 0] - P[bj + 0];
            float dy = P[bi + 1] - P[bj + 1];
            float dz = P[bi + 2] - P[bj + 2];
            dx -= Lx * rintf(dx * iLx);          // round-half-even, matches jnp.round
            dy -= Ly * rintf(dy * iLy);
            dz -= Lz * rintf(dz * iLz);
            float d2 = dx * dx + dy * dy + dz * dz;
            if (d2 >= 1e-12f && d2 <= XLIMF * XLIMF) {
                float d = sqrtf(d2);
                int b = (int)((d - B0F) * INV_W);
                b = b < 0 ? 0 : (b > NB - 1 ? NB - 1 : b);
                atomicAdd(&shh[b], 2u);          // (i,j) and (j,i)
            }
        }
    }
    __syncthreads();

    for (int b = tid; b < NB; b += 256) {
        unsigned int c = shh[b];
        if (c) atomicAdd(&ghist[r * NB + b], c);
    }
}

// ---------------- k2: normalize counts -> rdf ----------------
__global__ void k2_norm(const unsigned int* __restrict__ ghist,
                        const float* __restrict__ cell,
                        float* __restrict__ out) {
    int idx = blockIdx.x * blockDim.x + threadIdx.x;
    if (idx >= R_ * NB) return;
    int k = idx % NB;
    float vol = fabsf(cell[0] * cell[4] * cell[8]);
    float rho = (float)((double)T_ * N_ * N_) / vol;        // n_total / volume
    const float W = (XLIMF - B0F) / (float)NB;
    float lo = B0F + k * W;
    float hi = B0F + (k + 1) * W;
    float Z = rho * (4.0f / 3.0f) * 3.14159265358979323846f *
              (hi * hi * hi - lo * lo * lo);
    out[idx] = (float)ghist[idx] / Z;
}

// ---------------- k3: per-replica MAE via WMMA row-sum reduction ----------------
// maes[m] = 6/600 * sum_k |rdf[m,k]-gt[k]|.  One wave32, EXEC all-ones.
// B = all-ones makes every D column the A row-sum (B lane layout irrelevant).
__global__ __launch_bounds__(32) void k3_mae(const float* __restrict__ rdf,
                                             const float* __restrict__ gt,
                                             float* __restrict__ maes) {
    const int lane = threadIdx.x;
    v8f c = {};
#ifdef HAVE_WMMA_F32X4
    // A is 16x4 f32: lanes 0-15 hold K0/K1 in vgpr0/1, lanes 16-31 hold K2/K3.
    const int m  = lane & 15;
    const int kb = (lane >> 4) << 1;
    v2f bones; bones.x = 1.0f; bones.y = 1.0f;
    for (int it = 0; it < NB / 4; ++it) {       // 150 chained WMMAs
        int k0 = it * 4 + kb;
        v2f a;
        a.x = fabsf(rdf[m * NB + k0]     - gt[k0]);
        a.y = fabsf(rdf[m * NB + k0 + 1] - gt[k0 + 1]);
        c = __builtin_amdgcn_wmma_f32_16x16x4_f32(false, a, false, bones,
                                                  (short)0, c, false, false);
    }
#else
    // Fallback: f16 A (16x32), f32 accumulate.  A layout per ISA 16-bit table:
    // lanes 0-15: elems 0-7 -> K0-7, elems 8-15 -> K16-23; lanes 16-31: +8.
    const int m   = lane & 15;
    const int khi = (lane >> 4) << 3;
    v16h bones;
    for (int e = 0; e < 16; ++e) bones[e] = (_Float16)1.0f;
    for (int it = 0; it < 19; ++it) {           // 19*32 = 608 >= 600 (pad 0)
        v16h a;
        for (int e = 0; e < 16; ++e) {
            int kl = (e < 8) ? e : (e + 8);
            int k  = it * 32 + khi + kl;
            float v = (k < NB) ? fabsf(rdf[m * NB + k] - gt[k]) : 0.0f;
            a[e] = (_Float16)v;
        }
        c = __builtin_amdgcn_wmma_f32_16x16x32_f16(false, a, false, bones,
                                                   (short)0, c, false, false);
    }
#endif
    // D layout: vgpr v, lanes 0-15 -> M=v, lanes 16-31 -> M=v+8 (any column).
    if (lane == 0)  { for (int v = 0; v < 8; ++v) maes[v]     = 0.01f * c[v]; }
    if (lane == 16) { for (int v = 0; v < 8; ++v) maes[8 + v] = 0.01f * c[v]; }
}

// ---------------- launcher ----------------
extern "C" void kernel_launch(void* const* d_in, const int* in_sizes, int n_in,
                              void* d_out, int out_size, void* d_ws, size_t ws_size,
                              hipStream_t stream) {
    (void)in_sizes; (void)n_in; (void)out_size; (void)ws_size;
    const float* pos  = (const float*)d_in[0];   // [T,R,N,3]
    const float* gt   = (const float*)d_in[1];   // [600]
    const float* cell = (const float*)d_in[2];   // [3,3]
    float* out = (float*)d_out;                  // rdf [16*600] ++ maes [16]
    unsigned int* ghist = (unsigned int*)d_ws;   // 16*600 u32 counters

    k0_zero<<<(R_ * NB + 255) / 256, 256, 0, stream>>>(ghist, R_ * NB);
    k1_hist<<<dim3(R_, T_ / FPB), 256, 0, stream>>>(pos, cell, ghist);
    k2_norm<<<(R_ * NB + 255) / 256, 256, 0, stream>>>(ghist, cell, out);
    k3_mae<<<1, 32, 0, stream>>>(out, gt, out + R_ * NB);
}
